// Attention_29850022707970
// MI455X (gfx1250) — matrix operand order repaired and checked by reading, not compiled, not run on hardware
//
#include <hip/hip_runtime.h>
#include <hip/hip_bf16.h>

// ---------------------------------------------------------------------------
// GPT-2 attention block for gfx1250 (MI455X): QKV GEMM -> flash attention ->
// output projection, all matrix math on v_wmma_f32_16x16x32_bf16.
// B=2, S=2048, D=1024, H=16, HD=64
// ---------------------------------------------------------------------------

#define BB   2
#define SS   2048
#define DD   1024
#define HH   16
#define HDD  64
#define N3D  3072

typedef __bf16 v16bf __attribute__((ext_vector_type(16)));
typedef __bf16 v8bf  __attribute__((ext_vector_type(8)));
typedef float  v8f   __attribute__((ext_vector_type(8)));

__device__ __forceinline__ __bf16 tobf(float f) { return (__bf16)f; }

__device__ __forceinline__ v16bf combine16(v8bf lo, v8bf hi) {
  v16bf r;
#pragma unroll
  for (int i = 0; i < 8; i++) { r[i] = lo[i]; r[i + 8] = hi[i]; }
  return r;
}

__device__ __forceinline__ v8f wmma_bf16(v16bf a, v16bf b, v8f c) {
  return __builtin_amdgcn_wmma_f32_16x16x32_bf16(
      /*neg_a=*/false, a, /*neg_b=*/false, b,
      /*c_mod=*/(short)0, c, /*reuse_a=*/false, /*reuse_b=*/false);
}

// ---------------------------------------------------------------------------
// Kernel 1: QKV projection. C[4096,3072] = X[4096,1024] @ W[1024,3072] + bias.
// 4 waves/block, 128x64 tile/block, 32x64 strip/wave (8 WMMA/K-step),
// K-steps of 32. All fragments preloaded into distinct regs so the 8 WMMAs
// issue back-to-back. Epilogue scatters bf16 into Q[b,h,s,hd], K[b,h,s,hd],
// and transposed V[b,h,hd,s].
// ---------------------------------------------------------------------------
__global__ __launch_bounds__(128) void qkv_kernel(
    const float* __restrict__ X, const float* __restrict__ W,
    const float* __restrict__ bias,
    __bf16* __restrict__ Qb, __bf16* __restrict__ Kb, __bf16* __restrict__ Vb) {
  __shared__ __align__(16) __bf16 As[128 * 32];  // [row][k]
  __shared__ __align__(16) __bf16 Bs[64 * 32];   // transposed: [n][k]

  const int t    = threadIdx.x;
  const int lane = t & 31;
  const int wv   = t >> 5;
  const int m0   = blockIdx.x * 128;
  const int n0   = blockIdx.y * 64;
  const int hi   = (lane >= 16) ? 1 : 0;
  const int l16  = lane & 15;
  const int klo  = hi ? 8 : 0;    // A-fragment K offset
  const int kblo = hi ? 16 : 0;   // unused; B fragment also uses klo

  v8f acc[2][4] = {};

  for (int k0 = 0; k0 < DD; k0 += 32) {
    // ---- stage A tile 128x32 (fp32 -> bf16), two 64-row passes ----
#pragma unroll
    for (int j = 0; j < 2; j++) {
      const int row = (t >> 1) + j * 64, cs = (t & 1) * 16;
      const float* src = X + (size_t)(m0 + row) * DD + k0 + cs;
      if (k0 + 32 < DD) __builtin_prefetch(src + 32, 0, 3);
      v8bf t0, t1;
#pragma unroll
      for (int i = 0; i < 8; i++) { t0[i] = tobf(src[i]); t1[i] = tobf(src[8 + i]); }
      *(v8bf*)&As[row * 32 + cs]     = t0;
      *(v8bf*)&As[row * 32 + cs + 8] = t1;
    }
    // ---- stage B tile 32x64 transposed -> Bs[n][k] ----
    {
      const int kk = t >> 2, ns = (t & 3) * 16;
      const float* src = W + (size_t)(k0 + kk) * N3D + n0 + ns;
      if (k0 + 32 < DD) __builtin_prefetch(src + (size_t)32 * N3D, 0, 3);
#pragma unroll
      for (int i = 0; i < 16; i++) Bs[(ns + i) * 32 + kk] = tobf(src[i]);
    }
    __syncthreads();

    // ---- preload all fragments into distinct regs, then 8 WMMAs ----
    v16bf a[2];
#pragma unroll
    for (int sm = 0; sm < 2; sm++) {
      const int arow = wv * 32 + sm * 16 + l16;
      a[sm] = combine16(*(const v8bf*)&As[arow * 32 + klo],
                        *(const v8bf*)&As[arow * 32 + 16 + klo]);
    }
    v16bf bf[4];
#pragma unroll
    for (int tt = 0; tt < 4; tt++) {
      const int bn = tt * 16 + l16;
      bf[tt] = combine16(*(const v8bf*)&Bs[bn * 32 + klo],
                         *(const v8bf*)&Bs[bn * 32 + 16 + klo]);
    }
#pragma unroll
    for (int sm = 0; sm < 2; sm++)
#pragma unroll
      for (int tt = 0; tt < 4; tt++)
        acc[sm][tt] = wmma_bf16(a[sm], bf[tt], acc[sm][tt]);
    __syncthreads();
  }

  // ---- epilogue: + bias, scatter into Q / K / V^T (bf16) ----
#pragma unroll
  for (int tt = 0; tt < 4; tt++) {
    const int n  = n0 + tt * 16 + l16;
    const float bv = bias[n];
    const int sel = n >> 10;        // 0:q 1:k 2:v
    const int nn  = n & 1023;
    const int h   = nn >> 6;
    const int hd  = nn & 63;
#pragma unroll
    for (int sm = 0; sm < 2; sm++) {
#pragma unroll
      for (int r = 0; r < 8; r++) {
        const int m = m0 + wv * 32 + sm * 16 + r + hi * 8;
        const int b = m >> 11;      // row / S
        const int s = m & 2047;     // row % S
        const __bf16 o = tobf(acc[sm][tt][r] + bv);
        if (sel == 0)
          Qb[(((size_t)(b * HH + h)) * SS + s) * HDD + hd] = o;
        else if (sel == 1)
          Kb[(((size_t)(b * HH + h)) * SS + s) * HDD + hd] = o;
        else
          Vb[(((size_t)(b * HH + h)) * HDD + hd) * SS + s] = o;
      }
    }
  }
}

// ---------------------------------------------------------------------------
// Kernel 2: flash attention. One wave per (b, h, 16-query block).
// ---------------------------------------------------------------------------
__global__ __launch_bounds__(128) void attn_kernel(
    const __bf16* __restrict__ Qb, const __bf16* __restrict__ Kb,
    const __bf16* __restrict__ Vb, __bf16* __restrict__ Ab) {
  __shared__ __align__(16) __bf16 Pl[4][16 * 32];  // per-wave P tile [m][k]

  const int t    = threadIdx.x;
  const int lane = t & 31;
  const int wv   = t >> 5;
  const int gw   = blockIdx.x * 4 + wv;
  const int qblk = gw & 127;          // S/16 = 128
  const int h    = (gw >> 7) & 15;
  const int b    = gw >> 11;
  const int q0   = qblk * 16;
  const int hi   = (lane >= 16) ? 1 : 0;
  const int l16  = lane & 15;
  const int klo  = hi ? 8 : 0;        // 16-bit A/B fragment K offset (K=32 halves)
  const int kvlo = hi ? 16 : 0;       // unused; V fragment also uses klo

  const __bf16* qbase = Qb + ((size_t)(b * HH + h) * SS + q0) * HDD;
  const __bf16* kbh   = Kb + ((size_t)(b * HH + h) * SS) * HDD;
  const __bf16* vbh   = Vb + ((size_t)(b * HH + h) * HDD) * SS;

  // Q fragments: A-layout 16x32 for hd 0..31 and 32..63
  v16bf qa[2];
#pragma unroll
  for (int u = 0; u < 2; u++) {
    const __bf16* p = qbase + (size_t)l16 * HDD + u * 32 + klo;
    qa[u] = combine16(*(const v8bf*)p, *(const v8bf*)(p + 16));
  }

  v8f o[4] = {};
  float mstate[8], lstate[8];
#pragma unroll
  for (int r = 0; r < 8; r++) { mstate[r] = -1e30f; lstate[r] = 0.f; }

  const int nblocks = (q0 + 16 + 31) >> 5;   // causal: keys 0 .. q0+15
  for (int kb = 0; kb < nblocks; kb++) {
    const int k0 = kb * 32;

    // ---- preload K fragments (2 tiles x 2 K-halves) and V fragments ----
    v16bf kf[2][2];
#pragma unroll
    for (int tt = 0; tt < 2; tt++)
#pragma unroll
      for (int u = 0; u < 2; u++) {
        const __bf16* p = kbh + (size_t)(k0 + tt * 16 + l16) * HDD + u * 32 + klo;
        kf[tt][u] = combine16(*(const v8bf*)p, *(const v8bf*)(p + 16));
      }
    v16bf vf[4];
#pragma unroll
    for (int u = 0; u < 4; u++) {
      const __bf16* vp = vbh + (size_t)(u * 16 + l16) * SS + k0 + klo;
      vf[u] = combine16(*(const v8bf*)vp, *(const v8bf*)(vp + 16));
    }

    // ---- scores: two 16x16 tiles over this 32-key block ----
    float st[2][8];
#pragma unroll
    for (int tt = 0; tt < 2; tt++) {
      v8f s = {};
      s = wmma_bf16(qa[0], kf[tt][0], s);
      s = wmma_bf16(qa[1], kf[tt][1], s);
      const int ks = k0 + tt * 16 + l16;
#pragma unroll
      for (int r = 0; r < 8; r++) {
        const int qs = q0 + r + hi * 8;
        const float v = s[r] * 0.125f;              // 1/sqrt(64)
        st[tt][r] = (ks > qs) ? -10000.f : v;       // causal masked_bias
      }
    }

    // ---- online softmax (row = VGPR index within 16-lane half) ----
#pragma unroll
    for (int r = 0; r < 8; r++) {
      float mx = fmaxf(st[0][r], st[1][r]);
#pragma unroll
      for (int msk = 1; msk < 16; msk <<= 1) mx = fmaxf(mx, __shfl_xor(mx, msk, 32));
      const float newm  = fmaxf(mstate[r], mx);
      const float alpha = __expf(mstate[r] - newm);
      const float p0 = __expf(st[0][r] - newm);
      const float p1 = __expf(st[1][r] - newm);
      float rs = p0 + p1;
#pragma unroll
      for (int msk = 1; msk < 16; msk <<= 1) rs += __shfl_xor(rs, msk, 32);
      lstate[r] = lstate[r] * alpha + rs;
      mstate[r] = newm;
#pragma unroll
      for (int u = 0; u < 4; u++) o[u][r] *= alpha;
      const int m = r + hi * 8;
      Pl[wv][m * 32 + l16]      = tobf(p0);
      Pl[wv][m * 32 + 16 + l16] = tobf(p1);
    }

    // ---- P (A-layout via LDS) @ V ----
    const __bf16* pp = &Pl[wv][l16 * 32 + klo];
    v16bf pf = combine16(*(const v8bf*)pp, *(const v8bf*)(pp + 16));
#pragma unroll
    for (int u = 0; u < 4; u++) o[u] = wmma_bf16(pf, vf[u], o[u]);
  }

  // ---- normalize and write merged-head bf16 output [b, s, h*64+hd] ----
#pragma unroll
  for (int u = 0; u < 4; u++) {
#pragma unroll
    for (int r = 0; r < 8; r++) {
      const int m = r + hi * 8;
      const float val = o[u][r] / lstate[r];
      Ab[((size_t)(b * SS + q0 + m)) * DD + h * HDD + u * 16 + l16] = tobf(val);
    }
  }
}

// ---------------------------------------------------------------------------
// Kernel 3: output projection. out[4096,1024] = A[4096,1024] @ W[1024,1024]+b
// A is already bf16, so its LDS staging is a pure global->LDS copy: use the
// CDNA5 async data path (global_load_async_to_lds_b128 + s_wait_asynccnt)
// instead of a VGPR round-trip. W converted at staging; f32 output.
// ---------------------------------------------------------------------------
__global__ __launch_bounds__(128) void proj_kernel(
    const __bf16* __restrict__ Ab, const float* __restrict__ W,
    const float* __restrict__ bias, float* __restrict__ out) {
  __shared__ __align__(16) __bf16 As[128 * 32];
  __shared__ __align__(16) __bf16 Bs[64 * 32];   // transposed [n][k]

  const int t    = threadIdx.x;
  const int lane = t & 31;
  const int wv   = t >> 5;
  const int m0   = blockIdx.x * 128;
  const int n0   = blockIdx.y * 64;
  const int hi   = (lane >= 16) ? 1 : 0;
  const int l16  = lane & 15;
  const int klo  = hi ? 8 : 0;
  const int kblo = hi ? 16 : 0;

  v8f acc[2][4] = {};

  for (int k0 = 0; k0 < DD; k0 += 32) {
    // ---- async stage A tile 128x32 bf16 straight into LDS ----
#pragma unroll
    for (int j = 0; j < 2; j++) {
      const int row = (t >> 1) + j * 64, cs = (t & 1) * 16;
      const __bf16* src = Ab + (size_t)(m0 + row) * DD + k0 + cs;
      const unsigned lds0 = (unsigned)(size_t)&As[row * 32 + cs];
      const unsigned lds1 = lds0 + 16u;
      const unsigned long long ga0 = (unsigned long long)(size_t)src;
      const unsigned long long ga1 = ga0 + 16ull;
      asm volatile("global_load_async_to_lds_b128 %0, %1, off"
                   :: "v"(lds0), "v"(ga0) : "memory");
      asm volatile("global_load_async_to_lds_b128 %0, %1, off"
                   :: "v"(lds1), "v"(ga1) : "memory");
    }
    // ---- stage B tile 32x64 transposed (fp32 -> bf16) ----
    {
      const int kk = t >> 2, ns = (t & 3) * 16;
      const float* src = W + (size_t)(k0 + kk) * DD + n0 + ns;
      if (k0 + 32 < DD) __builtin_prefetch(src + (size_t)32 * DD, 0, 3);
#pragma unroll
      for (int i = 0; i < 16; i++) Bs[(ns + i) * 32 + kk] = tobf(src[i]);
    }
    asm volatile("s_wait_asynccnt 0" ::: "memory");
    __syncthreads();

    v16bf a[2];
#pragma unroll
    for (int sm = 0; sm < 2; sm++) {
      const int arow = wv * 32 + sm * 16 + l16;
      a[sm] = combine16(*(const v8bf*)&As[arow * 32 + klo],
                        *(const v8bf*)&As[arow * 32 + 16 + klo]);
    }
    v16bf bf[4];
#pragma unroll
    for (int tt = 0; tt < 4; tt++) {
      const int bn = tt * 16 + l16;
      bf[tt] = combine16(*(const v8bf*)&Bs[bn * 32 + klo],
                         *(const v8bf*)&Bs[bn * 32 + 16 + klo]);
    }
#pragma unroll
    for (int sm = 0; sm < 2; sm++)
#pragma unroll
      for (int tt = 0; tt < 4; tt++)
        acc[sm][tt] = wmma_bf16(a[sm], bf[tt], acc[sm][tt]);
    __syncthreads();
  }

#pragma unroll
  for (int tt = 0; tt < 4; tt++) {
    const int n = n0 + tt * 16 + l16;
    const float bv = bias[n];
#pragma unroll
    for (int sm = 0; sm < 2; sm++)
#pragma unroll
      for (int r = 0; r < 8; r++) {
        const int m = m0 + wv * 32 + sm * 16 + r + hi * 8;
        out[(size_t)m * DD + n] = acc[sm][tt][r] + bv;
      }
  }
}

// ---------------------------------------------------------------------------
extern "C" void kernel_launch(void* const* d_in, const int* in_sizes, int n_in,
                              void* d_out, int out_size, void* d_ws, size_t ws_size,
                              hipStream_t stream) {
  (void)in_sizes; (void)n_in; (void)out_size; (void)ws_size;

  const float* X     = (const float*)d_in[0];   // [B,S,D]
  const float* Wqkv  = (const float*)d_in[1];   // [D,3D]
  const float* bqkv  = (const float*)d_in[2];   // [3D]
  const float* Wproj = (const float*)d_in[3];   // [D,D]
  const float* bproj = (const float*)d_in[4];   // [D]
  float* out = (float*)d_out;                   // [B,S,D]

  const size_t per = (size_t)BB * HH * SS * HDD;  // 4M bf16 elements = 8 MB
  __bf16* Qb = (__bf16*)d_ws;
  __bf16* Kb = Qb + per;
  __bf16* Vb = Kb + per;                          // stored transposed [b,h,hd,s]
  __bf16* Ab = Vb + per;                          // [b,s,D] merged heads

  qkv_kernel<<<dim3(32, 48), 128, 0, stream>>>(X, Wqkv, bqkv, Qb, Kb, Vb);
  attn_kernel<<<dim3(1024), 128, 0, stream>>>(Qb, Kb, Vb, Ab);
  proj_kernel<<<dim3(32, 16), 128, 0, stream>>>(Ab, Wproj, bproj, out);
}
